// TacotronDecoder_53833120088273
// MI455X (gfx1250) — compile-verified, matrix operand order checked by reading, not run on hardware
//
#include <hip/hip_runtime.h>

// ---------------- constants ----------------
#define NWG   24          // persistent workgroups
#define BLK   256         // threads per workgroup (8 waves)
#define NTH   (NWG*BLK)

#define B_    32
#define TDEC  400
#define TENC  300
#define NMEL  80
#define PRE   256
#define ENCD  512
#define ARNN  512
#define DRNN  512
#define ATTD  128
#define LOCF  32
#define LOCK  31

typedef __attribute__((ext_vector_type(16))) _Float16 v16h;
typedef __attribute__((ext_vector_type(8)))  float    v8f;
typedef unsigned int v4u __attribute__((ext_vector_type(4)));
typedef int          v8i __attribute__((ext_vector_type(8)));
typedef int          v4i __attribute__((ext_vector_type(4)));

union Frag { v16h v; _Float16 h[16]; };

// ---------------- workspace layout (bytes) ----------------
constexpr size_t al256(size_t x){ return (x + 255) & ~size_t(255); }
constexpr size_t O_BAR   = 0;                                     // 2 ints (barrier)
constexpr size_t O_WP0H  = 256;                                   // [256][96]  f16 (padded K)
constexpr size_t O_WP1H  = al256(O_WP0H + 256*96*2);              // [256][256] f16
constexpr size_t O_WP2H  = al256(O_WP1H + 256*256*2);             // [256][256] f16
constexpr size_t O_WIHA  = al256(O_WP2H + 256*256*2);             // [1536][768] f16
constexpr size_t O_WHHA  = al256(O_WIHA + 1536*768*2);            // [1536][512] f16
constexpr size_t O_WIHD  = al256(O_WHHA + 1536*512*2);            // [1536][1024] f16
constexpr size_t O_WHHD  = al256(O_WIHD + 1536*1024*2);           // [1536][512] f16
constexpr size_t O_WQH   = al256(O_WHHD + 1536*512*2);            // [128][512] f16
constexpr size_t O_WOUTH = al256(O_WQH  + 128*512*2);             // [80][1024] f16
constexpr size_t O_PMEM  = al256(O_WOUTH + 80*1024*2);            // [32*300][128] f32
// ---- zero-initialized state block ----
constexpr size_t O_STATE = al256(O_PMEM + (size_t)B_*TENC*ATTD*4);
constexpr size_t O_AH    = O_STATE;                               // [32][512] f32
constexpr size_t O_DH    = O_AH  + B_*ARNN*4;                     // [32][512] f32
constexpr size_t O_AW    = O_DH  + B_*DRNN*4;                     // [32][300] f32
constexpr size_t O_AWS   = O_AW  + B_*TENC*4;                     // [32][300] f32
constexpr size_t O_XC    = O_AWS + B_*TENC*4;                     // [32][768]  f16  [x | ctx]
constexpr size_t O_ADC   = O_XC  + B_*768*2;                      // [32][1024] f16  [ah | ctx]
constexpr size_t O_DDC   = O_ADC + B_*1024*2;                     // [32][1024] f16  [dh | ctx]
constexpr size_t O_STATE_END = O_DDC + B_*1024*2;
// ---- scratch ----
constexpr size_t O_GHA   = al256(O_STATE_END);                    // [32][1536] f32
constexpr size_t O_GHD   = al256(O_GHA + B_*1536*4);              // [32][1536] f32
constexpr size_t O_GI    = al256(O_GHD + B_*1536*4);              // [32][1536] f32
constexpr size_t O_Q     = al256(O_GI  + B_*1536*4);              // [32][128]  f32
constexpr size_t O_E     = al256(O_Q   + B_*ATTD*4);              // [32][300]  f32
constexpr size_t O_END   = al256(O_E   + B_*TENC*4);

// ---------------- small helpers ----------------
__device__ __forceinline__ float sigm(float x){ return 1.f/(1.f+__expf(-x)); }

// deterministic dropout mask (p=0.5, inverted scaling) -- hash of (step,layer,idx)
__device__ __forceinline__ float dropmask(unsigned t, unsigned layer, unsigned idx){
  unsigned s = (t*3u + layer)*2654435761u ^ (idx*2246822519u) ^ 0x9e3779b9u;
  s ^= s >> 16; s *= 2654435761u; s ^= s >> 13; s *= 3266489917u; s ^= s >> 16;
  return (s & 1u) ? 2.0f : 0.0f;
}

// grid-wide sense-reversing barrier (persistent kernel; NWG co-resident WGs)
__device__ __forceinline__ void gbar(int* bar){
  __threadfence();          // make this thread's prior stores device-visible
  __syncthreads();
  if (threadIdx.x == 0){
    int* cnt = bar; int* gen = bar + 1;
    int g = __atomic_load_n(gen, __ATOMIC_ACQUIRE);
    int v = __atomic_add_fetch(cnt, 1, __ATOMIC_ACQ_REL);
    if (v == NWG){
      __atomic_store_n(cnt, 0, __ATOMIC_RELAXED);
      __atomic_store_n(gen, g + 1, __ATOMIC_RELEASE);
    } else {
      while (__atomic_load_n(gen, __ATOMIC_ACQUIRE) == g) __builtin_amdgcn_s_sleep(2);
    }
  }
  __syncthreads();
}

// ---------------- Tensor Data Mover staging ----------------
// One TDM descriptor load: copy a 2D tile (tile_w x tile_h, 4-byte elements,
// row stride `stride` elems) from global `gptr` into LDS at byte offset lds_off.
// D# layout per CDNA5 ISA 8.3/8.4 (group0: count/lds/global/type, group1: dims).
__device__ __forceinline__ void tdm_load_2d(unsigned lds_off, const void* gptr,
                                            unsigned tile_w, unsigned tile_h,
                                            unsigned stride){
  unsigned long long ga = (unsigned long long)gptr;
  v4u g0;
  g0[0] = 1u;                                           // count=1, user descriptor
  g0[1] = lds_off;                                      // lds_addr (bytes)
  g0[2] = (unsigned)(ga & 0xffffffffu);                 // global_addr[31:0]
  g0[3] = (unsigned)((ga >> 32) & 0x1ffffffu) | (2u << 30); // global_addr[56:32], type=2
  v8i g1;
  g1[0] = (int)(2u << 16);                              // wg_mask=0, data_size=2 (4 bytes)
  g1[1] = (int)((stride & 0xffffu) << 16);              // tensor_dim0[15:0]  (= stride)
  g1[2] = (int)((stride >> 16) & 0xffffu)
        | (int)((tile_h & 0xffffu) << 16);              // tensor_dim0[31:16], tensor_dim1[15:0]
  g1[3] = (int)((tile_h >> 16) & 0xffffu)
        | (int)((tile_w & 0xffffu) << 16);              // tensor_dim1[31:16], tile_dim0
  g1[4] = (int)(tile_h & 0xffffu);                      // tile_dim1 (tile_dim2=0)
  g1[5] = (int)stride;                                  // tensor_dim0_stride[31:0]
  g1[6] = 0;                                            // stride[47:32]=0, dim1_stride lo=0
  g1[7] = 0;
  v4i z4 = {0,0,0,0};
#if defined(__clang_major__) && (__clang_major__ >= 23)
  v8i z8 = {0,0,0,0,0,0,0,0};
  __builtin_amdgcn_tensor_load_to_lds(g0, g1, z4, z4, z8, 0);
#else
  __builtin_amdgcn_tensor_load_to_lds(g0, g1, z4, z4, 0);
#endif
}

// ---------------- WMMA fragment loaders (ISA 7.12.2 layouts) ----------------
__device__ __forceinline__ v16h load_a(const _Float16* X, int ldx, int mbase, int k0, int lane){
  Frag a;
  const _Float16* Xr = X + (mbase + (lane & 15)) * ldx;
  int kh = k0 + ((lane >> 4) << 3);
#pragma unroll
  for (int i = 0; i < 8; ++i){
    int k = kh + ((i >> 2) << 4) + ((i & 3) << 1);
    a.h[2*i]   = Xr[k];
    a.h[2*i+1] = Xr[k+1];
  }
  return a.v;
}
__device__ __forceinline__ v16h load_b(const _Float16* W, int ldw, int nbase, int k0, int lane){
  Frag b;
  const _Float16* Wr = W + (size_t)(nbase + (lane & 15)) * ldw;
  int kh = k0 + ((lane >> 4) << 3);
#pragma unroll
  for (int i = 0; i < 8; ++i){
    int k = kh + ((i >> 2) << 4) + ((i & 3) << 1);
    b.h[2*i]   = Wr[k];
    b.h[2*i+1] = Wr[k+1];
  }
  return b.v;
}

// one 16x16 f32 tile of  C = X[32,K] @ W.T  (W: [N][K] f16, L2-resident)
__device__ __forceinline__ v8f gemm_tile(const _Float16* Xl, int ldx, const _Float16* W, int ldw,
                                         int mbase, int ncol, int K, int lane){
  v8f c = {};
  for (int k0 = 0; k0 < K; k0 += 32){
    __builtin_prefetch(W + (size_t)(ncol + (lane & 15)) * ldw + k0 + 256, 0, 3); // near-scope
    v16h a = load_a(Xl, ldx, mbase, k0, lane);
    v16h b = load_b(W, ldw, ncol, k0, lane);
    c = __builtin_amdgcn_wmma_f32_16x16x32_f16(false, a, false, b, (short)0, c, false, false);
  }
  return c;
}

// C/D layout: VGPR r: lanes 0-15 -> M=r, lanes 16-31 -> M=r+8; N=lane&15
__device__ __forceinline__ void store_c(float* out, int ldo, int mbase, int nbase,
                                        const v8f& c, int lane){
  int n = nbase + (lane & 15), mo = (lane >> 4) * 8;
#pragma unroll
  for (int r = 0; r < 8; ++r) out[(size_t)(mbase + mo + r) * ldo + n] = c[r];
}

// ---------------- pre-pass kernels ----------------
__global__ void k_f16cvt(_Float16* dst, const float* src, int n){
  int i = blockIdx.x * blockDim.x + threadIdx.x;
  if (i < n) dst[i] = (_Float16)src[i];
}
__global__ void k_f16cvt_pad(_Float16* dst, const float* src, int rows, int kin, int kout){
  int i = blockIdx.x * blockDim.x + threadIdx.x;
  if (i < rows * kout){
    int r = i / kout, k = i - r * kout;
    dst[i] = (_Float16)((k < kin) ? src[r * kin + k] : 0.f);
  }
}
__global__ void k_pmem(float* pmem, const float* enc, const float* Wm){
  __shared__ float erow[ENCD];
  int row = blockIdx.x;
  for (int k = threadIdx.x; k < ENCD; k += 128) erow[k] = enc[(size_t)row * ENCD + k];
  __syncthreads();
  int d = threadIdx.x;
  const float* wr = Wm + (size_t)d * ENCD;
  float acc = 0.f;
  for (int k = 0; k < ENCD; ++k) acc += erow[k] * wr[k];
  pmem[(size_t)row * ATTD + d] = acc;
}

// GRU gate fusion; writes f32 state h and f16 mirror hh (stride 1024, cols 0..511)
__device__ __forceinline__ void gru_combine(const float* gi, const float* gh,
                                            const float* bi, const float* bh,
                                            float* h, _Float16* hh, int gtid){
  for (int idx = gtid; idx < 32 * 512; idx += NTH){
    int b = idx >> 9, c = idx & 511;
    const float* gib = gi + b * 1536;
    const float* ghb = gh + b * 1536;
    float r  = sigm(gib[c]        + bi[c]        + ghb[c]        + bh[c]);
    float z  = sigm(gib[c + 512]  + bi[c + 512]  + ghb[c + 512]  + bh[c + 512]);
    float hn = ghb[c + 1024] + bh[c + 1024];
    float nn = tanhf(gib[c + 1024] + bi[c + 1024] + r * hn);
    float hv = (1.f - z) * nn + z * h[idx];
    h[idx] = hv;
    hh[b * 1024 + c] = (_Float16)hv;
  }
}

// ---------------- persistent decoder ----------------
__global__ __launch_bounds__(BLK) void k_decoder(
    char* __restrict__ ws,
    const float* __restrict__ enc, const float* __restrict__ mels, const int* __restrict__ lengths,
    const float* __restrict__ bia, const float* __restrict__ bha,
    const float* __restrict__ bid, const float* __restrict__ bhd,
    const float* __restrict__ kloc, const float* __restrict__ wl, const float* __restrict__ vatt,
    const float* __restrict__ bout, const float* __restrict__ wgate, const float* __restrict__ bgate,
    float* __restrict__ out)
{
  __shared__ __align__(16) unsigned char smem[65536];
  _Float16* Xl = (_Float16*)smem;
  const unsigned lds_base = (unsigned)(size_t)(void*)smem;   // LDS byte offset for TDM

  const int tid = threadIdx.x, wg = blockIdx.x;
  const int lane = tid & 31, w = tid >> 5;
  const int gtid = wg * BLK + tid;

  int* bar = (int*)(ws + O_BAR);
  const _Float16* Wp0h  = (const _Float16*)(ws + O_WP0H);
  const _Float16* Wp1h  = (const _Float16*)(ws + O_WP1H);
  const _Float16* Wp2h  = (const _Float16*)(ws + O_WP2H);
  const _Float16* Wihah = (const _Float16*)(ws + O_WIHA);
  const _Float16* Whhah = (const _Float16*)(ws + O_WHHA);
  const _Float16* Wihdh = (const _Float16*)(ws + O_WIHD);
  const _Float16* Whhdh = (const _Float16*)(ws + O_WHHD);
  const _Float16* Wqh   = (const _Float16*)(ws + O_WQH);
  const _Float16* Wouth = (const _Float16*)(ws + O_WOUTH);
  const float* pmem = (const float*)(ws + O_PMEM);
  float* ah   = (float*)(ws + O_AH);
  float* dh   = (float*)(ws + O_DH);
  float* aw   = (float*)(ws + O_AW);
  float* aws  = (float*)(ws + O_AWS);
  _Float16* xc  = (_Float16*)(ws + O_XC);    // [32][768]  = [x | ctx]
  _Float16* adc = (_Float16*)(ws + O_ADC);   // [32][1024] = [ah | ctx]
  _Float16* ddc = (_Float16*)(ws + O_DDC);   // [32][1024] = [dh | ctx]
  float* gha  = (float*)(ws + O_GHA);
  float* ghd  = (float*)(ws + O_GHD);
  float* gi   = (float*)(ws + O_GI);
  float* qbuf = (float*)(ws + O_Q);
  float* ebuf = (float*)(ws + O_E);

  float* mel_out   = out;                        // [B][80][400]
  float* gate_out  = out + (size_t)B_*NMEL*TDEC; // [B][400]
  float* align_out = gate_out + (size_t)B_*TDEC; // [B][400][300]

  for (int t = 0; t < TDEC; ++t){
    // ============ P0: prenet (WG0)  ||  gh_a / gh_d (WG1..23) ============
    if (wg == 0){
      _Float16* bufA = Xl;             // [32][256] region
      _Float16* bufB = Xl + 32*256;    // [32][256] region
      for (int i = tid; i < 32*96; i += BLK){
        int m = i / 96, k = i - m*96;
        float v = (t > 0 && k < NMEL) ? mels[((size_t)m*TDEC + (t-1))*NMEL + k] : 0.f;
        bufA[i] = (_Float16)v;
      }
      __syncthreads();
      for (int ti = w; ti < 32; ti += 8){              // layer0 -> bufB
        int mt = ti & 1, nb = (ti >> 1) * 16;
        v8f c = gemm_tile(bufA, 96, Wp0h, 96, mt*16, nb, 96, lane);
        int n = nb + (lane & 15), mo = (lane >> 4) * 8;
#pragma unroll
        for (int r = 0; r < 8; ++r){
          int m = mt*16 + mo + r;
          float v = fmaxf(c[r], 0.f) * dropmask(t, 0, m*PRE + n);
          bufB[m*PRE + n] = (_Float16)v;
        }
      }
      __syncthreads();
      for (int ti = w; ti < 32; ti += 8){              // layer1 -> bufA
        int mt = ti & 1, nb = (ti >> 1) * 16;
        v8f c = gemm_tile(bufB, PRE, Wp1h, PRE, mt*16, nb, PRE, lane);
        int n = nb + (lane & 15), mo = (lane >> 4) * 8;
#pragma unroll
        for (int r = 0; r < 8; ++r){
          int m = mt*16 + mo + r;
          float v = fmaxf(c[r], 0.f) * dropmask(t, 1, m*PRE + n);
          bufA[m*PRE + n] = (_Float16)v;
        }
      }
      __syncthreads();
      for (int ti = w; ti < 32; ti += 8){              // layer2 -> xc (f16, cols 0..255)
        int mt = ti & 1, nb = (ti >> 1) * 16;
        v8f c = gemm_tile(bufA, PRE, Wp2h, PRE, mt*16, nb, PRE, lane);
        int n = nb + (lane & 15), mo = (lane >> 4) * 8;
#pragma unroll
        for (int r = 0; r < 8; ++r){
          int m = mt*16 + mo + r;
          xc[m*768 + n] = (_Float16)(fmaxf(c[r], 0.f) * dropmask(t, 2, m*PRE + n));
        }
      }
    } else {
      // 48 jobs: {gh_a: 24 col-blocks of 64, gh_d: 24 col-blocks}
      for (int job = wg - 1; job < 48; job += NWG - 1){
        int which = job / 24, cb = job % 24;
        const _Float16* hsrc = which ? ddc : adc;      // h in cols 0..511, stride 1024
        const _Float16* Wmat = which ? Whhdh : Whhah;
        float*          dst  = which ? ghd : gha;
        if (w == 0){                                    // TDM: 32 x 512 f16, stride 1024 f16
          tdm_load_2d(lds_base, hsrc, 256, 32, 512);
          __builtin_amdgcn_s_wait_tensorcnt(0);
        }
        __syncthreads();
        int mt = w & 1, nb = cb*64 + (w >> 1) * 16;
        v8f c = gemm_tile(Xl, 512, Wmat, 512, mt*16, nb, 512, lane);
        store_c(dst, 1536, mt*16, nb, c, lane);
        __syncthreads();
      }
    }
    gbar(bar);

    // ============ P1: gi_a = [x,ctx] @ W_ih_a.T  (K=768, N=1536, all WGs) ============
    if (w == 0){                                        // TDM: whole xc, contiguous 48KB
      tdm_load_2d(lds_base, xc, 12288, 1, 12288);
      __builtin_amdgcn_s_wait_tensorcnt(0);
    }
    __syncthreads();
    {
      int mt = w & 1, nb = wg*64 + (w >> 1) * 16;
      v8f c = gemm_tile(Xl, 768, Wihah, 768, mt*16, nb, 768, lane);
      store_c(gi, 1536, mt*16, nb, c, lane);
    }
    gbar(bar);

    // ============ P1b: attention GRU gates -> ah (+ f16 mirror in adc) ============
    gru_combine(gi, gha, bia, bha, ah, adc, gtid);
    gbar(bar);

    // ============ P2: q = ah @ W_q.T  (2 WGs, 16 tiles) ============
    if (wg < 2){
      if (w == 0){
        tdm_load_2d(lds_base, adc, 256, 32, 512);      // ah cols of adc
        __builtin_amdgcn_s_wait_tensorcnt(0);
      }
      __syncthreads();
      int wi = wg*8 + w, mt = wi & 1, nb = (wi >> 1) * 16;
      v8f c = gemm_tile(Xl, 512, Wqh, 512, mt*16, nb, 512, lane);
      store_c(qbuf, ATTD, mt*16, nb, c, lane);
    }
    gbar(bar);

    // ============ P2b: fused location-conv + proj + energy ============
    {
      float*    Klds = (float*)smem;            // 32*2*31 = 1984 f32
      float*    Wlld = Klds + 1984;             // 128*32  = 4096 f32
      float*    vld  = Wlld + 4096;             // 128 f32
      _Float16* awl  = (_Float16*)(vld + 128);  // 9600 f16
      _Float16* awsl = awl + 9600;              // 9600 f16
      for (int i = tid; i < 1984; i += BLK) Klds[i] = kloc[i];
      for (int i = tid; i < 4096; i += BLK) Wlld[i] = wl[i];
      for (int i = tid; i < 128;  i += BLK) vld[i]  = vatt[i];
      for (int i = tid; i < B_*TENC; i += BLK){ awl[i] = (_Float16)aw[i]; awsl[i] = (_Float16)aws[i]; }
      __syncthreads();
      for (int item = gtid; item < B_*TENC; item += NTH){
        int b = item / TENC, tt = item - b * TENC;
        float fo[LOCF];
        for (int f = 0; f < LOCF; ++f){
          float s = 0.f;
          const float* k0r = Klds + (f*2 + 0)*LOCK;
          const float* k1r = Klds + (f*2 + 1)*LOCK;
          for (int k = 0; k < LOCK; ++k){
            int p = tt + k - 15;
            if (p >= 0 && p < TENC){
              s += k0r[k] * (float)awl[b*TENC + p] + k1r[k] * (float)awsl[b*TENC + p];
            }
          }
          fo[f] = s;
        }
        const float* qb = qbuf + b * ATTD;
        const float* pm = pmem + (size_t)item * ATTD;
        float eacc = 0.f;
        for (int d = 0; d < ATTD; ++d){
          float s = qb[d] + pm[d];
          const float* wr = Wlld + d * LOCF;
          for (int f = 0; f < LOCF; ++f) s += wr[f] * fo[f];
          eacc += vld[d] * tanhf(s);
        }
        ebuf[item] = (tt < lengths[b]) ? eacc : -1e9f;
      }
    }
    gbar(bar);

    // ============ P3: masked softmax per batch row (one wave each) ============
    {
      int gwave = wg*8 + w;
      if (gwave < B_){
        int b = gwave;
        const float* eb = ebuf + b*TENC;
        float* awb = aw + b*TENC; float* awsb = aws + b*TENC;
        float mx = -1e30f;
        for (int tt = lane; tt < TENC; tt += 32) mx = fmaxf(mx, eb[tt]);
        for (int o = 16; o >= 1; o >>= 1) mx = fmaxf(mx, __shfl_xor(mx, o, 32));
        float sum = 0.f;
        for (int tt = lane; tt < TENC; tt += 32){
          float ex = __expf(eb[tt] - mx); awb[tt] = ex; sum += ex;
        }
        for (int o = 16; o >= 1; o >>= 1) sum += __shfl_xor(sum, o, 32);
        float inv = 1.f / sum;
        for (int tt = lane; tt < TENC; tt += 32){
          float a = awb[tt] * inv;
          awb[tt] = a; awsb[tt] += a;
          align_out[((size_t)b*TDEC + t)*TENC + tt] = a;
        }
      }
    }
    gbar(bar);

    // ============ P4: ctx[b][d] = sum_t aw[b][t] * enc[b][t][d]  (f16 mirrors) ============
    for (int p = gtid; p < B_*ENCD; p += NTH){
      int b = p >> 9, d = p & 511;
      const float* awb = aw + b*TENC;
      const float* eb  = enc + (size_t)b*TENC*ENCD + d;
      float a = 0.f;
      for (int tt = 0; tt < TENC; ++tt) a += awb[tt] * eb[(size_t)tt*ENCD];
      _Float16 hv = (_Float16)a;
      xc [b*768  + 256 + d] = hv;
      adc[b*1024 + 512 + d] = hv;
      ddc[b*1024 + 512 + d] = hv;
    }
    gbar(bar);

    // ============ P5: gi_d = [ah,ctx] @ W_ih_d.T  (K=1024, all WGs) ============
    if (w == 0){                                        // TDM: whole adc, contiguous 64KB
      tdm_load_2d(lds_base, adc, 16384, 1, 16384);
      __builtin_amdgcn_s_wait_tensorcnt(0);
    }
    __syncthreads();
    {
      int mt = w & 1, nb = wg*64 + (w >> 1) * 16;
      v8f c = gemm_tile(Xl, 1024, Wihdh, 1024, mt*16, nb, 1024, lane);
      store_c(gi, 1536, mt*16, nb, c, lane);
    }
    gbar(bar);

    // ============ P5b: decoder GRU gates -> dh (+ f16 mirror in ddc) ============
    gru_combine(gi, ghd, bid, bhd, dh, ddc, gtid);
    gbar(bar);

    // ============ P6: mel = [dh,ctx]@W_out.T + b  ||  gate ============
    if (wg < 3){
      if (w == 0){                                      // TDM: whole ddc, contiguous 64KB
        tdm_load_2d(lds_base, ddc, 16384, 1, 16384);
        __builtin_amdgcn_s_wait_tensorcnt(0);
      }
      __syncthreads();
      if (wg < 2){
        int ti = wg*8 + w;
        if (ti < 10){
          int mt = ti & 1, nb = (ti >> 1) * 16;        // 5 N-tiles cover 80 cols
          v8f c = gemm_tile(Xl, 1024, Wouth, 1024, mt*16, nb, 1024, lane);
          int n = nb + (lane & 15), mo = (lane >> 4) * 8;
#pragma unroll
          for (int r = 0; r < 8; ++r){
            int b = mt*16 + mo + r;
            mel_out[((size_t)b*NMEL + n)*TDEC + t] = c[r] + bout[n];
          }
        }
      } else { // wg == 2: gate projection, one wave per 4 rows
        for (int rep = 0; rep < 4; ++rep){
          int b = w*4 + rep;
          float a = 0.f;
          for (int k = lane; k < 1024; k += 32) a += (float)Xl[b*1024 + k] * wgate[k];
          for (int o = 16; o >= 1; o >>= 1) a += __shfl_xor(a, o, 32);
          if (lane == 0) gate_out[(size_t)b*TDEC + t] = a + bgate[0];
        }
      }
    }
    gbar(bar);
  }
}

// ---------------- host launcher ----------------
extern "C" void kernel_launch(void* const* d_in, const int* in_sizes, int n_in,
                              void* d_out, int out_size, void* d_ws, size_t ws_size,
                              hipStream_t stream){
  (void)in_sizes; (void)n_in; (void)out_size;
  if (ws_size < O_END) return;

  const float* enc  = (const float*)d_in[0];
  const float* mels = (const float*)d_in[1];
  const int*   len  = (const int*)d_in[2];
  char* ws = (char*)d_ws;

  hipMemsetAsync(ws + O_BAR, 0, 256, stream);
  hipMemsetAsync(ws + O_STATE, 0, O_STATE_END - O_STATE, stream);

  auto cvt = [&](size_t off, int idx, int n){
    k_f16cvt<<<(n + 255)/256, 256, 0, stream>>>((_Float16*)(ws + off), (const float*)d_in[idx], n);
  };
  k_f16cvt_pad<<<(256*96 + 255)/256, 256, 0, stream>>>((_Float16*)(ws + O_WP0H),
                                                       (const float*)d_in[3], 256, 80, 96);
  cvt(O_WP1H, 4, 256*256);
  cvt(O_WP2H, 5, 256*256);
  cvt(O_WIHA, 6, 1536*768);
  cvt(O_WHHA, 7, 1536*512);
  cvt(O_WIHD, 10, 1536*1024);
  cvt(O_WHHD, 11, 1536*512);
  cvt(O_WQH,  14, 128*512);
  cvt(O_WOUTH,19, 80*1024);

  k_pmem<<<B_*TENC, 128, 0, stream>>>((float*)(ws + O_PMEM), enc, (const float*)d_in[15]);

  k_decoder<<<NWG, BLK, 0, stream>>>(
      ws, enc, mels, len,
      (const float*)d_in[8],  (const float*)d_in[9],
      (const float*)d_in[12], (const float*)d_in[13],
      (const float*)d_in[16], (const float*)d_in[17], (const float*)d_in[18],
      (const float*)d_in[20], (const float*)d_in[21], (const float*)d_in[22],
      (float*)d_out);
}
